// DeformableAttention_4518305595598
// MI455X (gfx1250) — matrix operand order repaired and checked by reading, not compiled, not run on hardware
//
#include <hip/hip_runtime.h>
#include <hip/hip_bf16.h>

#define BB 8
#define CC 256
#define NN 1024
#define NH 8
#define HD 32
#define SR 4
static constexpr float SCALE_QK = 0.17677669529663687f; // 32^-0.5

typedef __attribute__((ext_vector_type(16))) _Float16 v16h;
typedef __attribute__((ext_vector_type(8)))  _Float16 v8h;
typedef __attribute__((ext_vector_type(8)))  float    v8f;

__device__ __forceinline__ v8f wmma16(v16h a, v16h b, v8f c) {
  // D = A(16x32 f16) * B(32x16 f16) + C(16x16 f32)
  return __builtin_amdgcn_wmma_f32_16x16x32_f16(false, a, false, b, (short)0, c,
                                                false, false);
}

// A fragment (16x32): rowptr = this lane's row; 16 contiguous halves along K at k0+16*hi.
__device__ __forceinline__ v16h load_a_row(const _Float16* rowptr, int k0, int hi) {
  union { v16h v; v8h h[2]; } u;
  u.h[0] = *(const v8h*)(rowptr + k0 + (hi << 4));
  u.h[1] = *(const v8h*)(rowptr + k0 + 8 + (hi << 4));
  return u.v;
}
// B fragment (32x16): p = &mem[col][k_base]; 16 contiguous halves along K.
__device__ __forceinline__ v16h load_b_contig(const _Float16* p) {
  union { v16h v; v8h h[2]; } u;
  u.h[0] = *(const v8h*)(p);
  u.h[1] = *(const v8h*)(p + 8);
  return u.v;
}
__device__ __forceinline__ float redmax16(float v) {
  #pragma unroll
  for (int m = 1; m < 16; m <<= 1) v = fmaxf(v, __shfl_xor(v, m, 16));
  return v;
}
__device__ __forceinline__ float redsum16(float v) {
  #pragma unroll
  for (int m = 1; m < 16; m <<= 1) v += __shfl_xor(v, m, 16);
  return v;
}

// ---------------- Kernel 0a: x [B,C,N] f32 -> Xt [B,N,C] f16 ----------------
__global__ void k_transpose(const float* __restrict__ x, _Float16* __restrict__ xt) {
  int idx = blockIdx.x * blockDim.x + threadIdx.x;
  if (idx >= BB * NN * CC) return;
  int c = idx % CC;
  int n = (idx / CC) % NN;
  int b = idx / (CC * NN);
  xt[idx] = (_Float16)x[(b * CC + c) * NN + n];
}

// ------- Kernel 0b: stack+convert weights [Wq;Wk;Wv;Wp] -> f16 [1024][256] -------
__global__ void k_cvtw(const float* __restrict__ Wq, const float* __restrict__ Wk,
                       const float* __restrict__ Wv, const float* __restrict__ Wp,
                       _Float16* __restrict__ Wh) {
  int idx = blockIdx.x * blockDim.x + threadIdx.x; // over 1024*256
  if (idx >= 4 * CC * CC) return;
  int row = idx / CC;
  int c   = idx % CC;
  const float* src =
      (row < 256) ? (Wq + (size_t)row * CC)
    : (row < 512) ? (Wk + (size_t)(row - 256) * CC)
    : (row < 768) ? (Wv + (size_t)(row - 512) * CC)
                  : (Wp + (size_t)(row - 768) * CC);
  Wh[idx] = (_Float16)src[c];
}

// ---------------- Kernel 1: gumbel-softmax gates (channel 0) ----------------
__global__ void k_gates(const _Float16* __restrict__ xt,
                        const float* __restrict__ Wsq, const float* __restrict__ bsq,
                        const float* __restrict__ Wsk, const float* __restrict__ bsk,
                        const float* __restrict__ gq, const float* __restrict__ gk,
                        float* __restrict__ sq, float* __restrict__ sk) {
  __shared__ float ws[2 * SR * CC];
  for (int i = threadIdx.x; i < SR * CC; i += blockDim.x) {
    ws[i] = Wsq[i];
    ws[SR * CC + i] = Wsk[i];
  }
  __syncthreads();
  int pix = blockIdx.x * blockDim.x + threadIdx.x; // [0, B*N)
  int b = pix / NN, n = pix % NN;
  const _Float16* xp = xt + (size_t)pix * CC;
  float acc[2 * SR];
  #pragma unroll
  for (int j = 0; j < 2 * SR; ++j) acc[j] = 0.f;
  for (int c = 0; c < CC; ++c) {
    float xv = (float)xp[c];
    #pragma unroll
    for (int j = 0; j < 2 * SR; ++j) acc[j] += ws[j * CC + c] * xv;
  }
  float zq[SR], zk[SR];
  #pragma unroll
  for (int j = 0; j < SR; ++j) {
    zq[j] = acc[j] + bsq[j] + gq[(b * SR + j) * NN + n];
    zk[j] = acc[SR + j] + bsk[j] + gk[(b * SR + j) * NN + n];
  }
  float mq = zq[0], mk = zk[0];
  #pragma unroll
  for (int j = 1; j < SR; ++j) { mq = fmaxf(mq, zq[j]); mk = fmaxf(mk, zk[j]); }
  float dq = 0.f, dk = 0.f;
  #pragma unroll
  for (int j = 0; j < SR; ++j) { dq += __expf(zq[j] - mq); dk += __expf(zk[j] - mk); }
  sq[pix] = __expf(zq[0] - mq) / dq;
  sk[pix] = __expf(zk[0] - mk) / dk;
}

// ------------- Kernel 2: fused QKV projection GEMM (WMMA) -------------
// One wave per 16-pixel x 64-outchannel tile (4 accumulators share one A frag).
// All 5 fragment loads per K-step are issued before the 4 WMMAs (MLP).
__global__ void k_qkv(const _Float16* __restrict__ xt, const _Float16* __restrict__ Wh,
                      const float* __restrict__ bq, const float* __restrict__ bk,
                      const float* __restrict__ bv,
                      const float* __restrict__ sq, const float* __restrict__ sk,
                      _Float16* __restrict__ qh, _Float16* __restrict__ kh,
                      _Float16* __restrict__ vh, _Float16* __restrict__ vt) {
  int lane = threadIdx.x;
  int l = lane & 15, hi = lane >> 4;
  int bid = blockIdx.x;
  int ot = bid % 12;             // 64-wide output-channel tile over 768
  int nt = (bid / 12) % 64;      // pixel tile within batch
  int b  = bid / (12 * 64);
  int cg = ot * 64;              // stacked channel base
  int which = cg >> 8;           // 0=q 1=k 2=v (64 | 256, never straddles)
  int co = cg & 255;             // channel base within one matrix
  int n0 = nt * 16;

  const float* bias = (which == 0) ? bq : (which == 1) ? bk : bv;
  const _Float16* arow = xt + ((size_t)(b * NN + n0 + l)) * CC;
  const _Float16* w0 = Wh + (size_t)(cg + 0 * 16 + l) * CC + hi * 16;
  const _Float16* w1 = Wh + (size_t)(cg + 1 * 16 + l) * CC + hi * 16;
  const _Float16* w2 = Wh + (size_t)(cg + 2 * 16 + l) * CC + hi * 16;
  const _Float16* w3 = Wh + (size_t)(cg + 3 * 16 + l) * CC + hi * 16;

  v8f acc0 = {}, acc1 = {}, acc2 = {}, acc3 = {};
  #pragma unroll
  for (int k0 = 0; k0 < CC; k0 += 32) {
    v16h a   = load_a_row(arow, k0, hi);
    v16h bf0 = load_b_contig(w0 + k0);
    v16h bf1 = load_b_contig(w1 + k0);
    v16h bf2 = load_b_contig(w2 + k0);
    v16h bf3 = load_b_contig(w3 + k0);
    acc0 = wmma16(a, bf0, acc0);
    acc1 = wmma16(a, bf1, acc1);
    acc2 = wmma16(a, bf2, acc2);
    acc3 = wmma16(a, bf3, acc3);
  }

  v8f accs[4] = {acc0, acc1, acc2, acc3};
  #pragma unroll
  for (int t = 0; t < 4; ++t) {
    int col = co + t * 16 + l;
    float bb = bias[col];
    int h = col >> 5, d = col & 31;
    #pragma unroll
    for (int r = 0; r < 8; ++r) {
      int n = n0 + r + hi * 8;
      float val = accs[t][r] + bb;
      if (which == 0)      val *= sq[b * NN + n] * SCALE_QK;
      else if (which == 1) val *= sk[b * NN + n];
      _Float16 hv = (_Float16)val;
      size_t hidx = ((size_t)(b * NH + h) * NN + n) * HD + d;
      if (which == 0)      qh[hidx] = hv;
      else if (which == 1) kh[hidx] = hv;
      else {
        vh[hidx] = hv;
        vt[((size_t)(b * NH + h) * HD + d) * NN + n] = hv; // d-major copy for PV
      }
    }
  }
}

// ------------- Kernel 3: flash attention + gating (WMMA) -------------
// One wave per (b, head, 16-query tile); streams 32 keys/iter.
__global__ void k_attn(const _Float16* __restrict__ qh, const _Float16* __restrict__ kh,
                       const _Float16* __restrict__ vt, const _Float16* __restrict__ vh,
                       const float* __restrict__ sqg, _Float16* __restrict__ G) {
  __shared__ __align__(16) _Float16 Pl[16 * 32];
  int lane = threadIdx.x;
  int l = lane & 15, hi = lane >> 4;
  int bid = blockIdx.x;
  int qt = bid & 63;
  int h  = (bid >> 6) & 7;
  int b  = bid >> 9;
  int q0 = qt * 16;

  const _Float16* Qb = qh + (size_t)(b * NH + h) * NN * HD;
  const _Float16* Kb = kh + (size_t)(b * NH + h) * NN * HD;
  const _Float16* Vt = vt + (size_t)(b * NH + h) * HD * NN;
  const _Float16* Vh = vh + (size_t)(b * NH + h) * NN * HD;

  v16h qa = load_a_row(Qb + (size_t)(q0 + l) * HD, 0, hi);

  v8f o0 = {}, o1 = {};
  float mrow[8], lrow[8];
  #pragma unroll
  for (int r = 0; r < 8; ++r) { mrow[r] = -INFINITY; lrow[r] = 0.f; }

  for (int key0 = 0; key0 < NN; key0 += 32) {
    v16h kb0 = load_b_contig(Kb + (size_t)(key0 + l) * HD + hi * 16);
    v16h kb1 = load_b_contig(Kb + (size_t)(key0 + 16 + l) * HD + hi * 16);
    v8f zc = {};
    v8f s0 = wmma16(qa, kb0, zc);
    v8f s1 = wmma16(qa, kb1, zc);

    // V fragments don't depend on P: issue their loads under the softmax VALU.
    v16h vb0 = load_b_contig(Vt + (size_t)l * NN + key0 + hi * 16);
    v16h vb1 = load_b_contig(Vt + (size_t)(16 + l) * NN + key0 + hi * 16);

    // Prefetch next chunk's K and V rows (global_prefetch_b8).
    if (key0 + 32 < NN) {
      __builtin_prefetch(Kb + (size_t)(key0 + 32 + l) * HD + hi * 16, 0, 3);
      __builtin_prefetch(Kb + (size_t)(key0 + 48 + l) * HD + hi * 16, 0, 3);
      __builtin_prefetch(Vt + (size_t)l * NN + key0 + 32 + hi * 16, 0, 3);
      __builtin_prefetch(Vt + (size_t)(16 + l) * NN + key0 + 32 + hi * 16, 0, 3);
    }

    float p0a[8], p1a[8];
    #pragma unroll
    for (int r = 0; r < 8; ++r) {
      float tm = redmax16(fmaxf(s0[r], s1[r]));
      float mnew = fmaxf(mrow[r], tm);
      float alpha = __expf(mrow[r] - mnew);
      mrow[r] = mnew;
      float p0 = __expf(s0[r] - mnew);
      float p1 = __expf(s1[r] - mnew);
      p0a[r] = p0; p1a[r] = p1;
      lrow[r] = lrow[r] * alpha + redsum16(p0 + p1);
      o0[r] *= alpha;
      o1[r] *= alpha;
    }

    __syncthreads();               // previous iteration's Pl reads done
    #pragma unroll
    for (int r = 0; r < 8; ++r) {
      int row = r + hi * 8;
      Pl[row * 32 + l]      = (_Float16)p0a[r];
      Pl[row * 32 + 16 + l] = (_Float16)p1a[r];
    }
    __syncthreads();               // Pl stores visible

    v16h pa = load_a_row(Pl + l * 32, 0, hi);
    o0 = wmma16(pa, vb0, o0);
    o1 = wmma16(pa, vb1, o1);
  }

  #pragma unroll
  for (int r = 0; r < 8; ++r) {
    int n = q0 + r + hi * 8;
    float inv = 1.f / lrow[r];
    float s = sqg[b * NN + n];
    float out0 = o0[r] * inv;
    float out1 = o1[r] * inv;
    float g0 = s * out0 + (1.f - s) * (float)Vh[(size_t)n * HD + l];
    float g1 = s * out1 + (1.f - s) * (float)Vh[(size_t)n * HD + 16 + l];
    size_t gbase = ((size_t)(b * NN + n)) * CC + h * HD;
    G[gbase + l]      = (_Float16)g0;
    G[gbase + 16 + l] = (_Float16)g1;
  }
}

// ------------- Kernel 4: output projection GEMM (WMMA) -------------
// One wave per 16-pixel x 64-outchannel tile; Wp = rows 768..1023 of Wh.
__global__ void k_proj(const _Float16* __restrict__ G, const _Float16* __restrict__ Wh,
                       const float* __restrict__ bp, float* __restrict__ out) {
  int lane = threadIdx.x;
  int l = lane & 15, hi = lane >> 4;
  int bid = blockIdx.x;
  int ct = bid & 3;
  int nt = (bid >> 2) & 63;
  int b  = bid >> 8;
  int n0 = nt * 16, co = ct * 64;

  const _Float16* arow = G + ((size_t)(b * NN + n0 + l)) * CC;
  const _Float16* wp   = Wh + (size_t)768 * CC;
  const _Float16* w0 = wp + (size_t)(co + 0 * 16 + l) * CC + hi * 16;
  const _Float16* w1 = wp + (size_t)(co + 1 * 16 + l) * CC + hi * 16;
  const _Float16* w2 = wp + (size_t)(co + 2 * 16 + l) * CC + hi * 16;
  const _Float16* w3 = wp + (size_t)(co + 3 * 16 + l) * CC + hi * 16;

  v8f acc0 = {}, acc1 = {}, acc2 = {}, acc3 = {};
  #pragma unroll
  for (int k0 = 0; k0 < CC; k0 += 32) {
    v16h a   = load_a_row(arow, k0, hi);
    v16h bf0 = load_b_contig(w0 + k0);
    v16h bf1 = load_b_contig(w1 + k0);
    v16h bf2 = load_b_contig(w2 + k0);
    v16h bf3 = load_b_contig(w3 + k0);
    acc0 = wmma16(a, bf0, acc0);
    acc1 = wmma16(a, bf1, acc1);
    acc2 = wmma16(a, bf2, acc2);
    acc3 = wmma16(a, bf3, acc3);
  }
  v8f accs[4] = {acc0, acc1, acc2, acc3};
  #pragma unroll
  for (int t = 0; t < 4; ++t) {
    int col = co + t * 16 + l;
    float bb = bp[col];
    #pragma unroll
    for (int r = 0; r < 8; ++r) {
      int n = n0 + r + hi * 8;
      out[((size_t)(b * CC + col)) * NN + n] = accs[t][r] + bb;
    }
  }
}

extern "C" void kernel_launch(void* const* d_in, const int* in_sizes, int n_in,
                              void* d_out, int out_size, void* d_ws, size_t ws_size,
                              hipStream_t stream) {
  const float* x   = (const float*)d_in[0];
  const float* gq  = (const float*)d_in[1];
  const float* gk  = (const float*)d_in[2];
  const float* Wsq = (const float*)d_in[3];
  const float* bsq = (const float*)d_in[4];
  const float* Wsk = (const float*)d_in[5];
  const float* bsk = (const float*)d_in[6];
  const float* Wq  = (const float*)d_in[7];
  const float* bq  = (const float*)d_in[8];
  const float* Wk  = (const float*)d_in[9];
  const float* bk  = (const float*)d_in[10];
  const float* Wv  = (const float*)d_in[11];
  const float* bv  = (const float*)d_in[12];
  const float* Wp  = (const float*)d_in[13];
  const float* bp  = (const float*)d_in[14];
  float* out = (float*)d_out;

  // Workspace carve-out
  size_t off = 0;
  auto carve = [&](size_t bytes) -> char* {
    char* p = (char*)d_ws + off;
    off += (bytes + 255) & ~(size_t)255;
    return p;
  };
  const size_t n_bnc = (size_t)BB * NN * CC;     // 2,097,152 elements
  _Float16* xt = (_Float16*)carve(n_bnc * 2);    // [B,N,C]
  _Float16* qh = (_Float16*)carve(n_bnc * 2);    // [B,H,N,HD]
  _Float16* kh = (_Float16*)carve(n_bnc * 2);
  _Float16* vh = (_Float16*)carve(n_bnc * 2);
  _Float16* vt = (_Float16*)carve(n_bnc * 2);    // [B,H,HD,N]
  _Float16* G  = (_Float16*)carve(n_bnc * 2);    // gated, [B,N,C]
  _Float16* Wh = (_Float16*)carve((size_t)4 * CC * CC * 2); // [Wq;Wk;Wv;Wp] f16
  float* sq = (float*)carve((size_t)BB * NN * 4);
  float* sk = (float*)carve((size_t)BB * NN * 4);

  k_cvtw<<<(4 * CC * CC + 255) / 256, 256, 0, stream>>>(Wq, Wk, Wv, Wp, Wh);
  k_transpose<<<(BB * NN * CC + 255) / 256, 256, 0, stream>>>(x, xt);
  k_gates<<<(BB * NN) / 64, 64, 0, stream>>>(xt, Wsq, bsq, Wsk, bsk, gq, gk, sq, sk);
  k_qkv<<<BB * 64 * 12, 32, 0, stream>>>(xt, Wh, bq, bk, bv, sq, sk, qh, kh, vh, vt);
  k_attn<<<BB * NH * 64, 32, 0, stream>>>(qh, kh, vt, vh, sq, G);
  k_proj<<<BB * 64 * 4, 32, 0, stream>>>(G, Wh, bp, out);
}
